// SelfAttentionBlock_65481071406807
// MI455X (gfx1250) — matrix operand order repaired and checked by reading, not compiled, not run on hardware
//
#include <hip/hip_runtime.h>
#include <math.h>

typedef __bf16 bf16_t;
typedef __attribute__((ext_vector_type(16))) __bf16 v16bf;
typedef __attribute__((ext_vector_type(8)))  float  v8f;

#define B_   4
#define N_   2048
#define D_   1024
#define BN   (B_ * N_)          // 8192 token rows
#define QKVC (3 * D_)           // 3072 qkv columns

// Wave-level macro-tile: 32 (M) x 64 (N) => 2x4 grid of 16x16 WMMA tiles.
#define MI 2
#define NJ 4
#define TM (MI * 16)            // 32
#define TN (NJ * 16)            // 64

// ---------- WMMA fragment loads (contiguous-K order, same for A and B) ----------
// A 16x32 bf16: lane holds row (lane&15); half=lane>>4 selects K {0..15}/{16..31},
//   16 contiguous K values per lane (matches B fragment order)
__device__ __forceinline__ v16bf ld_frag_a(const bf16_t* rowk, int half) {
  union { uint4 u[2]; v16bf v; } u;
  u.u[0] = *reinterpret_cast<const uint4*>(rowk + half * 16);
  u.u[1] = *reinterpret_cast<const uint4*>(rowk + half * 16 + 8);
  return u.v;
}
// B 32x16 bf16: lane holds column (lane&15); half selects K {0..15}/{16..31},
// 16 contiguous K values per lane.
__device__ __forceinline__ v16bf ld_frag_b(const bf16_t* rowk, int half) {
  const bf16_t* p = rowk + half * 16;
  union { uint4 u[2]; v16bf v; } u;
  u.u[0] = *reinterpret_cast<const uint4*>(p);
  u.u[1] = *reinterpret_cast<const uint4*>(p + 8);
  return u.v;
}

// 32x64 macro-tile GEMM: C += A(32 x K) * B(64 x K)^T, both row-major K-contig.
// a_row/b_row already include the per-lane (lane&15) row offset.
__device__ __forceinline__ void gemm_tile_32x64(const bf16_t* a_row, size_t lda,
                                                const bf16_t* b_row, size_t ldb,
                                                int kdim, v8f acc[MI][NJ]) {
  const int half = (threadIdx.x >> 4) & 1;
  for (int k0 = 0; k0 < kdim; k0 += 32) {
    v16bf a[MI], b[NJ];
#pragma unroll
    for (int i = 0; i < MI; ++i)
      a[i] = ld_frag_a(a_row + (size_t)i * 16 * lda + k0, half);
#pragma unroll
    for (int j = 0; j < NJ; ++j)
      b[j] = ld_frag_b(b_row + (size_t)j * 16 * ldb + k0, half);
#pragma unroll
    for (int i = 0; i < MI; ++i)
#pragma unroll
      for (int j = 0; j < NJ; ++j)
        acc[i][j] = __builtin_amdgcn_wmma_f32_16x16x32_bf16(
            false, a[i], false, b[j], (short)0, acc[i][j], false, false);
  }
}

__device__ __forceinline__ void zero_acc(v8f acc[MI][NJ]) {
  const v8f z = {0.f, 0.f, 0.f, 0.f, 0.f, 0.f, 0.f, 0.f};
#pragma unroll
  for (int i = 0; i < MI; ++i)
#pragma unroll
    for (int j = 0; j < NJ; ++j) acc[i][j] = z;
}

// ---------- kernels ----------
__global__ void k_cvt(const float* __restrict__ s, bf16_t* __restrict__ d, int n) {
  int i = blockIdx.x * blockDim.x + threadIdx.x;
  if (i < n) d[i] = (bf16_t)s[i];
}

// QKV = X * Wqkv^T ; q,k -> qk[8192][2048] bf16 ; v -> vT[b][1024][2048] bf16
__global__ void k_qkv(const bf16_t* __restrict__ xb, const bf16_t* __restrict__ wb,
                      bf16_t* __restrict__ qk, bf16_t* __restrict__ vT) {
  int wave = (blockIdx.x * blockDim.x + threadIdx.x) >> 5;
  const int tilesX = QKVC / TN;  // 48
  int tr = wave / tilesX, tc = wave % tilesX;
  if (tr >= BN / TM) return;
  int lane = threadIdx.x & 31, l16 = lane & 15, half = lane >> 4;
  int row0 = tr * TM, col0 = tc * TN;
  const bf16_t* a_row = xb + (size_t)(row0 + l16) * D_;
  const bf16_t* b_row = wb + (size_t)(col0 + l16) * D_;
  v8f acc[MI][NJ];
  zero_acc(acc);
  gemm_tile_32x64(a_row, D_, b_row, D_, D_, acc);
#pragma unroll
  for (int j = 0; j < NJ; ++j) {
    bool isV = (col0 + j * 16 >= 2 * D_);  // 64-col tiles never straddle the V boundary
#pragma unroll
    for (int i = 0; i < MI; ++i) {
#pragma unroll
      for (int r = 0; r < 8; ++r) {
        int gr = row0 + i * 16 + r + half * 8;
        int gc = col0 + j * 16 + l16;
        bf16_t v = (bf16_t)acc[i][j][r];
        if (!isV) {
          qk[(size_t)gr * (2 * D_) + gc] = v;
        } else {
          int b = gr >> 11, n = gr & (N_ - 1);
          vT[((size_t)b * D_ + (gc - 2 * D_)) * N_ + n] = v;  // transposed V
        }
      }
    }
  }
}

// S[b][n][m] = (Q . K) / 8 , f32
__global__ void k_scores(const bf16_t* __restrict__ qk, float* __restrict__ S) {
  int wave = (blockIdx.x * blockDim.x + threadIdx.x) >> 5;
  const int tR = N_ / TM, tC = N_ / TN;  // 64, 32
  int b = wave / (tR * tC);
  if (b >= B_) return;
  int rem = wave % (tR * tC);
  int tr = rem / tC, tc = rem % tC;
  int lane = threadIdx.x & 31, l16 = lane & 15, half = lane >> 4;
  const bf16_t* a_row = qk + (size_t)(b * N_ + tr * TM + l16) * (2 * D_);        // Q
  const bf16_t* b_row = qk + (size_t)(b * N_ + tc * TN + l16) * (2 * D_) + D_;   // K
  v8f acc[MI][NJ];
  zero_acc(acc);
  gemm_tile_32x64(a_row, 2 * D_, b_row, 2 * D_, D_, acc);
#pragma unroll
  for (int i = 0; i < MI; ++i)
#pragma unroll
    for (int j = 0; j < NJ; ++j)
#pragma unroll
      for (int r = 0; r < 8; ++r)
        S[((size_t)b * N_ + tr * TM + i * 16 + r + half * 8) * N_ +
          tc * TN + j * 16 + l16] = acc[i][j][r] * 0.125f;
}

// row softmax over 2048 f32, bf16 probabilities written in-place (ldP = 4096 bf16)
__global__ void k_softmax(float* __restrict__ S) {
  float* x = S + (size_t)blockIdx.x * N_;
  int tid = threadIdx.x;  // 256 threads, 8 elems each
  float v[8];
#pragma unroll
  for (int i = 0; i < 8; ++i) v[i] = x[tid + i * 256];
  __shared__ float red[256];
  float m = -3.4e38f;
#pragma unroll
  for (int i = 0; i < 8; ++i) m = fmaxf(m, v[i]);
  red[tid] = m;
  __syncthreads();
  for (int s = 128; s > 0; s >>= 1) {
    if (tid < s) red[tid] = fmaxf(red[tid], red[tid + s]);
    __syncthreads();
  }
  m = red[0];
  __syncthreads();
  float sum = 0.f;
#pragma unroll
  for (int i = 0; i < 8; ++i) { v[i] = __expf(v[i] - m); sum += v[i]; }
  red[tid] = sum;
  __syncthreads();
  for (int s = 128; s > 0; s >>= 1) {
    if (tid < s) red[tid] += red[tid + s];
    __syncthreads();
  }
  float inv = 1.0f / red[0];
  // all f32 reads for this row happened before the syncs above -> safe in-place
  bf16_t* p = (bf16_t*)x;
#pragma unroll
  for (int i = 0; i < 8; ++i) p[tid + i * 256] = (bf16_t)(v[i] * inv);
}

// AO = P * V ; P rows bf16 (ld 4096), V supplied transposed so B-frag is contiguous
__global__ void k_av(const bf16_t* __restrict__ P, const bf16_t* __restrict__ vT,
                     bf16_t* __restrict__ AO) {
  int wave = (blockIdx.x * blockDim.x + threadIdx.x) >> 5;
  const int tilesX = D_ / TN;  // 16
  int tr = wave / tilesX, tc = wave % tilesX;
  if (tr >= BN / TM) return;
  int lane = threadIdx.x & 31, l16 = lane & 15, half = lane >> 4;
  int row0 = tr * TM, col0 = tc * TN;
  int b = row0 >> 11;  // 32-row tiles never straddle batches
  const bf16_t* a_row = P + (size_t)(row0 + l16) * (2 * N_);          // ldP = 4096
  const bf16_t* b_row = vT + ((size_t)b * D_ + col0 + l16) * N_;
  v8f acc[MI][NJ];
  zero_acc(acc);
  gemm_tile_32x64(a_row, 2 * N_, b_row, N_, N_, acc);
#pragma unroll
  for (int i = 0; i < MI; ++i)
#pragma unroll
    for (int j = 0; j < NJ; ++j)
#pragma unroll
      for (int r = 0; r < 8; ++r)
        AO[(size_t)(row0 + i * 16 + r + half * 8) * D_ + col0 + j * 16 + l16] =
            (bf16_t)acc[i][j][r];
}

// Y = AO * Wout^T + b_out , f32 out
__global__ void k_out(const bf16_t* __restrict__ AO, const bf16_t* __restrict__ wob,
                      const float* __restrict__ bias, float* __restrict__ out) {
  int wave = (blockIdx.x * blockDim.x + threadIdx.x) >> 5;
  const int tilesX = D_ / TN;  // 16
  int tr = wave / tilesX, tc = wave % tilesX;
  if (tr >= BN / TM) return;
  int lane = threadIdx.x & 31, l16 = lane & 15, half = lane >> 4;
  int row0 = tr * TM, col0 = tc * TN;
  const bf16_t* a_row = AO + (size_t)(row0 + l16) * D_;
  const bf16_t* b_row = wob + (size_t)(col0 + l16) * D_;
  v8f acc[MI][NJ];
  zero_acc(acc);
  gemm_tile_32x64(a_row, D_, b_row, D_, D_, acc);
#pragma unroll
  for (int i = 0; i < MI; ++i)
#pragma unroll
    for (int j = 0; j < NJ; ++j) {
      float bb = bias[col0 + j * 16 + l16];
#pragma unroll
      for (int r = 0; r < 8; ++r)
        out[(size_t)(row0 + i * 16 + r + half * 8) * D_ + col0 + j * 16 + l16] =
            acc[i][j][r] + bb;
    }
}

// ---------- launch ----------
extern "C" void kernel_launch(void* const* d_in, const int* in_sizes, int n_in,
                              void* d_out, int out_size, void* d_ws, size_t ws_size,
                              hipStream_t stream) {
  (void)in_sizes; (void)n_in; (void)out_size; (void)ws_size;
  const float* x     = (const float*)d_in[0];
  const float* w_qkv = (const float*)d_in[1];
  const float* w_out = (const float*)d_in[2];
  const float* b_out = (const float*)d_in[3];
  float* out = (float*)d_out;

  char* ws = (char*)d_ws;                              // bytes      (256-aligned)
  bf16_t* xb    = (bf16_t*)(ws + 0);                   //  16 MB  x in bf16
  bf16_t* wqkvb = (bf16_t*)(ws + 16777216);            //   6 MB
  bf16_t* woutb = (bf16_t*)(ws + 23068672);            //   2 MB
  bf16_t* qk    = (bf16_t*)(ws + 25165824);            //  32 MB  [8192][2048]
  bf16_t* vT    = (bf16_t*)(ws + 58720256);            //  16 MB  [4][1024][2048]
  float*  S     = (float*) (ws + 75497472);            //  64 MB  scores / P (in-place)
  bf16_t* AO    = (bf16_t*)(ws + 142606336);           //  16 MB  [8192][1024]

  k_cvt<<<(BN * D_ + 255) / 256, 256, 0, stream>>>(x, xb, BN * D_);
  k_cvt<<<(QKVC * D_ + 255) / 256, 256, 0, stream>>>(w_qkv, wqkvb, QKVC * D_);
  k_cvt<<<(D_ * D_ + 255) / 256, 256, 0, stream>>>(w_out, woutb, D_ * D_);

  {
    int tiles = (BN / TM) * (QKVC / TN);               // 256*48 = 12288 waves
    k_qkv<<<tiles / 8, 256, 0, stream>>>(xb, wqkvb, qk, vT);
  }
  {
    int tiles = B_ * (N_ / TM) * (N_ / TN);            // 4*64*32 = 8192 waves
    k_scores<<<tiles / 8, 256, 0, stream>>>(qk, S);
  }
  k_softmax<<<BN, 256, 0, stream>>>(S);
  {
    int tiles = (BN / TM) * (D_ / TN);                 // 256*16 = 4096 waves
    k_av<<<tiles / 8, 256, 0, stream>>>((const bf16_t*)S, vT, AO);
  }
  {
    int tiles = (BN / TM) * (D_ / TN);                 // 4096 waves
    k_out<<<tiles / 8, 256, 0, stream>>>(AO, woutb, b_out, out);
  }
}
